// BaseWindowAttention_62062277427844
// MI455X (gfx1250) — compile-verified, matrix operand order checked by reading, not run on hardware
//
#include <hip/hip_runtime.h>
#include <hip/hip_bf16.h>

typedef __attribute__((ext_vector_type(16))) _Float16 v16h;
typedef __attribute__((ext_vector_type(8)))  float    v8f;
typedef __attribute__((ext_vector_type(4)))  float    f32x4;
typedef __attribute__((ext_vector_type(4)))  unsigned int u32x4;

union Frag {
    v16h     v;
    _Float16 e[16];
    u32x4    q[2];
};

__device__ __forceinline__ v8f wmma_f16(v16h a, v16h b, v8f c) {
    // v_wmma_f32_16x16x32_f16: D = A(16x32 f16) * B(32x16 f16) + C(16x16 f32)
    return __builtin_amdgcn_wmma_f32_16x16x32_f16(false, a, false, b, (short)0, c, false, false);
}

// ---------------- small helper kernels ----------------

__global__ void cvt_f32_f16_kernel(const float* __restrict__ src,
                                   _Float16* __restrict__ dst, int n) {
    int i = blockIdx.x * 256 + threadIdx.x;
    if (i < n) dst[i] = (_Float16)src[i];
}

// bias64[i][j] = pos_embedding[jy-iy+7][jx-ix+7]   (rel = idx[j]-idx[i]+7)
__global__ void build_bias_kernel(const float* __restrict__ pos,
                                  float* __restrict__ bias) {
    int idx = blockIdx.x * 256 + threadIdx.x;   // 0..4095
    int i = idx >> 6, j = idx & 63;
    int iy = i >> 3, ix = i & 7, jy = j >> 3, jx = j & 7;
    bias[idx] = pos[(jy - iy + 7) * 15 + (jx - ix + 7)];
}

// ---------------- QKV projection GEMM ----------------
// qkv[m,o] = sum_c x[m,c] * w_qkv[o,c]   M=131072 N=768 K=256
// block = 4 waves, wave computes 32(M) x 128(N); grid (M/128, N/128)
// Two A-frags amortize each B-frag load: 16 wmma per ~24 vector loads.

__global__ __launch_bounds__(128)
void gemm_qkv_kernel(const float* __restrict__ x,
                     const _Float16* __restrict__ w,   // (768,256) f16 row-major
                     _Float16* __restrict__ qkv) {     // (131072,768) f16
    const int lane = threadIdx.x & 31;
    const int wave = threadIdx.x >> 5;
    const int l16  = lane & 15;
    const int hf   = lane >> 4;            // half-wave select
    const int m0   = blockIdx.x * 128 + wave * 32;
    const int n0   = blockIdx.y * 128;
    const int rowA = m0 + l16;
    const int rowB = m0 + 16 + l16;

    const v8f vzero = {0.f,0.f,0.f,0.f,0.f,0.f,0.f,0.f};
    v8f acc0[8], acc1[8];
    for (int t = 0; t < 8; ++t) { acc0[t] = vzero; acc1[t] = vzero; }

    for (int kk = 0; kk < 256; kk += 32) {
        // A fragments (f32 -> f16): lane holds row, halves [0..7]@k0, [8..15]@k0+16
        Frag a0, a1;
        {
            const float* ap = x + (size_t)rowA * 256 + kk + hf * 8;
            f32x4 f0 = *(const f32x4*)(ap);
            f32x4 f1 = *(const f32x4*)(ap + 4);
            f32x4 f2 = *(const f32x4*)(ap + 16);
            f32x4 f3 = *(const f32x4*)(ap + 20);
            for (int t = 0; t < 4; ++t) {
                a0.e[t]      = (_Float16)f0[t];
                a0.e[4 + t]  = (_Float16)f1[t];
                a0.e[8 + t]  = (_Float16)f2[t];
                a0.e[12 + t] = (_Float16)f3[t];
            }
        }
        {
            const float* ap = x + (size_t)rowB * 256 + kk + hf * 8;
            f32x4 f0 = *(const f32x4*)(ap);
            f32x4 f1 = *(const f32x4*)(ap + 4);
            f32x4 f2 = *(const f32x4*)(ap + 16);
            f32x4 f3 = *(const f32x4*)(ap + 20);
            for (int t = 0; t < 4; ++t) {
                a1.e[t]      = (_Float16)f0[t];
                a1.e[4 + t]  = (_Float16)f1[t];
                a1.e[8 + t]  = (_Float16)f2[t];
                a1.e[12 + t] = (_Float16)f3[t];
            }
        }
        for (int nt = 0; nt < 8; ++nt) {
            const _Float16* bp = w + (size_t)(n0 + nt * 16 + l16) * 256 + kk + hf * 8;
            Frag b;
            b.q[0] = *(const u32x4*)bp;
            b.q[1] = *(const u32x4*)(bp + 16);
            acc0[nt] = wmma_f16(a0.v, b.v, acc0[nt]);
            acc1[nt] = wmma_f16(a1.v, b.v, acc1[nt]);
        }
    }
    // C layout: VGPR e, lanes<16 -> M=e, lanes>=16 -> M=e+8; N = l16
    for (int nt = 0; nt < 8; ++nt)
        for (int e = 0; e < 8; ++e) {
            int oo = n0 + nt * 16 + l16;
            int mA = m0 + e + 8 * hf;
            int mB = m0 + 16 + e + 8 * hf;
            qkv[(size_t)mA * 768 + oo] = (_Float16)acc0[nt][e];
            qkv[(size_t)mB * 768 + oo] = (_Float16)acc1[nt][e];
        }
}

// ---------------- windowed attention ----------------
// one block (4 waves) per (b*l, head, window); q,k,v are 64x32 slices of qkv

__global__ __launch_bounds__(128)
void win_attn_kernel(const _Float16* __restrict__ qkv,
                     const float* __restrict__ bias,   // 64x64
                     _Float16* __restrict__ aoh) {     // (131072,256) f16
    __shared__ __align__(16) _Float16 sP[64 * 64];     // softmaxed probs, f16
    __shared__ __align__(16) _Float16 sVt[32 * 64];    // V transposed (c-major)

    const int tid  = threadIdx.x;
    const int lane = tid & 31, wave = tid >> 5;
    const int l16  = lane & 15, hf = lane >> 4;
    const int wh   = blockIdx.x;
    const int win  = wh & 255;
    const int h    = (wh >> 8) & 7;
    const int bl   = wh >> 11;
    const int wy = win >> 4, wx = win & 15;
    const int base_y = wy * 8, base_x = wx * 8;
    const size_t blbase = (size_t)bl * 16384;
    const int ho = h * 32;

    auto tok = [&](int i) -> size_t {   // within-window pos -> token index
        return blbase + (size_t)(base_y + (i >> 3)) * 128 + (base_x + (i & 7));
    };

    // cooperative transpose of V into LDS: sVt[c*64 + j] = v[j][c]
    {
        int j  = tid >> 1;
        int cb = (tid & 1) * 16;
        const _Float16* vp = qkv + tok(j) * 768 + 512 + ho + cb;
        Frag t;
        t.q[0] = *(const u32x4*)vp;
        t.q[1] = *(const u32x4*)(vp + 8);
        for (int c = 0; c < 16; ++c) sVt[(cb + c) * 64 + j] = t.e[c];
    }

    // S = q k^T  (wave handles rows 16*wave .. +16, all 64 cols)
    Frag aq;
    {
        int qi = wave * 16 + l16;
        const _Float16* qp = qkv + tok(qi) * 768 + ho + hf * 8;
        aq.q[0] = *(const u32x4*)qp;
        aq.q[1] = *(const u32x4*)(qp + 16);
    }
    const v8f vzero = {0.f,0.f,0.f,0.f,0.f,0.f,0.f,0.f};
    v8f s[4];
    for (int jt = 0; jt < 4; ++jt) {
        int kj = jt * 16 + l16;
        const _Float16* kp = qkv + tok(kj) * 768 + 256 + ho + hf * 8;
        Frag bk;
        bk.q[0] = *(const u32x4*)kp;
        bk.q[1] = *(const u32x4*)(kp + 16);
        s[jt] = wmma_f16(aq.v, bk.v, vzero);
    }
    const float SCALE = 0.17677669529663687f;  // 32^-0.5
    for (int jt = 0; jt < 4; ++jt)
        for (int e = 0; e < 8; ++e) {
            int i = wave * 16 + e + 8 * hf;
            int j = jt * 16 + l16;
            s[jt][e] = s[jt][e] * SCALE + bias[i * 64 + j];
        }

    // softmax over j: each row lives in one 16-lane half across 4 N-tiles
    for (int e = 0; e < 8; ++e) {
        float mx = fmaxf(fmaxf(s[0][e], s[1][e]), fmaxf(s[2][e], s[3][e]));
        for (int m = 1; m < 16; m <<= 1) mx = fmaxf(mx, __shfl_xor(mx, m, 32));
        float p0 = __expf(s[0][e] - mx);
        float p1 = __expf(s[1][e] - mx);
        float p2 = __expf(s[2][e] - mx);
        float p3 = __expf(s[3][e] - mx);
        float sum = (p0 + p1) + (p2 + p3);
        for (int m = 1; m < 16; m <<= 1) sum += __shfl_xor(sum, m, 32);
        float inv = 1.0f / sum;
        s[0][e] = p0 * inv; s[1][e] = p1 * inv;
        s[2][e] = p2 * inv; s[3][e] = p3 * inv;
    }

    // stash P (f16) row-major in LDS so it can be re-read in A-layout
    for (int jt = 0; jt < 4; ++jt)
        for (int e = 0; e < 8; ++e) {
            int i = wave * 16 + e + 8 * hf;
            int j = jt * 16 + l16;
            sP[i * 64 + j] = (_Float16)s[jt][e];
        }
    __syncthreads();

    // O = P @ V   (K=64 as two 32-chunks; N=32 as two 16-tiles)
    v8f o0 = vzero, o1 = vzero;
    for (int kc = 0; kc < 2; ++kc) {
        int r = wave * 16 + l16;
        const _Float16* pp = sP + r * 64 + hf * 8 + kc * 32;
        Frag ap;
        ap.q[0] = *(const u32x4*)pp;
        ap.q[1] = *(const u32x4*)(pp + 16);
        {
            const _Float16* vp = sVt + l16 * 64 + hf * 8 + kc * 32;
            Frag bv;
            bv.q[0] = *(const u32x4*)vp;
            bv.q[1] = *(const u32x4*)(vp + 16);
            o0 = wmma_f16(ap.v, bv.v, o0);
        }
        {
            const _Float16* vp = sVt + (16 + l16) * 64 + hf * 8 + kc * 32;
            Frag bv;
            bv.q[0] = *(const u32x4*)vp;
            bv.q[1] = *(const u32x4*)(vp + 16);
            o1 = wmma_f16(ap.v, bv.v, o1);
        }
    }
    for (int e = 0; e < 8; ++e) {
        int i = wave * 16 + e + 8 * hf;
        size_t tb = tok(i) * 256 + ho;
        aoh[tb + l16]      = (_Float16)o0[e];
        aoh[tb + 16 + l16] = (_Float16)o1[e];
    }
}

// ---------------- output projection GEMM (+bias) ----------------
// out[m,o] = sum_c aoh[m,c] * w_out[o,c] + b_out[o]   M=131072 N=256 K=256
// wave computes 32x128; grid (M/128, N/128)

__global__ __launch_bounds__(128)
void gemm_out_kernel(const _Float16* __restrict__ a,
                     const _Float16* __restrict__ w,
                     const float* __restrict__ bout,
                     float* __restrict__ out) {
    const int lane = threadIdx.x & 31;
    const int wave = threadIdx.x >> 5;
    const int l16  = lane & 15, hf = lane >> 4;
    const int m0   = blockIdx.x * 128 + wave * 32;
    const int n0   = blockIdx.y * 128;
    const int rowA = m0 + l16;
    const int rowB = m0 + 16 + l16;

    const v8f vzero = {0.f,0.f,0.f,0.f,0.f,0.f,0.f,0.f};
    v8f acc0[8], acc1[8];
    for (int t = 0; t < 8; ++t) { acc0[t] = vzero; acc1[t] = vzero; }

    for (int kk = 0; kk < 256; kk += 32) {
        Frag a0, a1;
        {
            const _Float16* apt = a + (size_t)rowA * 256 + kk + hf * 8;
            a0.q[0] = *(const u32x4*)apt;
            a0.q[1] = *(const u32x4*)(apt + 16);
        }
        {
            const _Float16* apt = a + (size_t)rowB * 256 + kk + hf * 8;
            a1.q[0] = *(const u32x4*)apt;
            a1.q[1] = *(const u32x4*)(apt + 16);
        }
        for (int nt = 0; nt < 8; ++nt) {
            const _Float16* bp = w + (size_t)(n0 + nt * 16 + l16) * 256 + kk + hf * 8;
            Frag b;
            b.q[0] = *(const u32x4*)bp;
            b.q[1] = *(const u32x4*)(bp + 16);
            acc0[nt] = wmma_f16(a0.v, b.v, acc0[nt]);
            acc1[nt] = wmma_f16(a1.v, b.v, acc1[nt]);
        }
    }
    for (int nt = 0; nt < 8; ++nt) {
        int oo = n0 + nt * 16 + l16;
        float bb = bout[oo];
        for (int e = 0; e < 8; ++e) {
            int mA = m0 + e + 8 * hf;
            int mB = m0 + 16 + e + 8 * hf;
            out[(size_t)mA * 256 + oo] = acc0[nt][e] + bb;
            out[(size_t)mB * 256 + oo] = acc1[nt][e] + bb;
        }
    }
}

// ---------------- launch ----------------

extern "C" void kernel_launch(void* const* d_in, const int* in_sizes, int n_in,
                              void* d_out, int out_size, void* d_ws, size_t ws_size,
                              hipStream_t stream) {
    const float* x    = (const float*)d_in[0];  // (2,4,128,128,256)
    const float* wqkv = (const float*)d_in[1];  // (768,256)
    const float* pos  = (const float*)d_in[2];  // (15,15)
    const float* wout = (const float*)d_in[3];  // (256,256)
    const float* bout = (const float*)d_in[4];  // (256,)
    float* out = (float*)d_out;

    // scratch layout
    char* ws = (char*)d_ws;
    _Float16* qkvh  = (_Float16*)ws;  ws += (size_t)131072 * 768 * 2;  // 201.3 MB
    _Float16* woh   = (_Float16*)ws;  ws += (size_t)768 * 256 * 2;
    _Float16* wouth = (_Float16*)ws;  ws += (size_t)256 * 256 * 2;
    float*    bias  = (float*)ws;     ws += (size_t)64 * 64 * 4;
    _Float16* aoh   = (_Float16*)ws;                                   // 67.1 MB

    cvt_f32_f16_kernel<<<(196608 + 255) / 256, 256, 0, stream>>>(wqkv, woh, 196608);
    cvt_f32_f16_kernel<<<(65536 + 255) / 256, 256, 0, stream>>>(wout, wouth, 65536);
    build_bias_kernel<<<16, 256, 0, stream>>>(pos, bias);
    gemm_qkv_kernel<<<dim3(1024, 6), 128, 0, stream>>>(x, woh, qkvh);
    win_attn_kernel<<<16384, 128, 0, stream>>>(qkvh, bias, aoh);
    gemm_out_kernel<<<dim3(1024, 2), 128, 0, stream>>>(aoh, wouth, bout, out);
}
